// EfficientMultiHeadAttention_21638045237272
// MI455X (gfx1250) — matrix operand order repaired and checked by reading, not compiled, not run on hardware
//
#include <hip/hip_runtime.h>

// ---------------------------------------------------------------------------
// EfficientMultiHeadAttention for MI455X (gfx1250, wave32, WMMA)
// B=2, S=2048, D=1024, H=16, hd=64
// ---------------------------------------------------------------------------

#define B_   2
#define S_   2048
#define D_   1024
#define H_   16
#define HD_  64
#define M_   (B_ * S_)          // 4096 rows of x
#define N3_  (3 * D_)           // 3072

typedef __attribute__((ext_vector_type(16))) __bf16 v16bf;
typedef __attribute__((ext_vector_type(8)))  float  v8f;

union V16U { v16bf v; float4 f[2]; };

static __device__ __forceinline__ __bf16 f2bf(float x) {
    unsigned int u = __builtin_bit_cast(unsigned int, x);
    unsigned int r = (u + 0x7FFFu + ((u >> 16) & 1u)) >> 16;
    unsigned short s = (unsigned short)r;
    return __builtin_bit_cast(__bf16, s);
}

// A fragment: 16x32 bf16 tile, row-major source with `stride` elements/row.
// lane m = lane&15 -> row m; k-groups at k0+half*8 and +16.
static __device__ __forceinline__ v16bf load_a16(const __bf16* base, int stride, int k0) {
    int lane = threadIdx.x & 31;
    int m    = lane & 15;
    int off  = (lane >> 4) << 3;
    const __bf16* p = base + m * stride + k0 + off;
    V16U u;
    u.f[0] = *(const float4*)(p);
    u.f[1] = *(const float4*)(p + 16);
    return u.v;
}

// B fragment: 32x16 bf16 (K x N); source stored "transposed" (row n holds
// column n of B contiguously): lane n reads k-groups at k0+half*8 and +16.
static __device__ __forceinline__ v16bf load_bt(const __bf16* bt, int stride, int n0, int k0) {
    int lane = threadIdx.x & 31;
    int n    = lane & 15;
    int off  = (lane >> 4) << 3;
    const __bf16* p = bt + (n0 + n) * stride + k0 + off;
    V16U u;
    u.f[0] = *(const float4*)(p);
    u.f[1] = *(const float4*)(p + 16);
    return u.v;
}

static __device__ __forceinline__ v8f wmma_bf16(v16bf a, v16bf b, v8f c) {
    return __builtin_amdgcn_wmma_f32_16x16x32_bf16(false, a, false, b, (short)0, c, false, false);
}

static __device__ __forceinline__ float rowmax16(float v) {
    v = fmaxf(v, __shfl_xor(v, 1, 16));
    v = fmaxf(v, __shfl_xor(v, 2, 16));
    v = fmaxf(v, __shfl_xor(v, 4, 16));
    v = fmaxf(v, __shfl_xor(v, 8, 16));
    return v;
}
static __device__ __forceinline__ float rowsum16(float v) {
    v += __shfl_xor(v, 1, 16);
    v += __shfl_xor(v, 2, 16);
    v += __shfl_xor(v, 4, 16);
    v += __shfl_xor(v, 8, 16);
    return v;
}

// ---------------------------------------------------------------------------
// Kernel 1: fp32 -> bf16 cast
// ---------------------------------------------------------------------------
__global__ void k_cast_bf16(const float* __restrict__ in, __bf16* __restrict__ out, int n) {
    int i = blockIdx.x * blockDim.x + threadIdx.x;
    if (i < n) out[i] = f2bf(in[i]);
}

// Kernel 2: fp32 [rows][cols] -> bf16 transposed [cols][rows]
__global__ void k_transpose_bf16(const float* __restrict__ in, __bf16* __restrict__ out,
                                 int rows, int cols) {
    int i = blockIdx.x * blockDim.x + threadIdx.x;
    if (i < rows * cols) {
        int c = i / rows;
        int r = i - c * rows;
        out[i] = f2bf(in[r * cols + c]);
    }
}

// ---------------------------------------------------------------------------
// Kernel 3: fused QKV projection.  C = X(bf16) @ Wqkv + b, scattered into
// per-head Q [B,H,S,hd], K [B,H,S,hd], Vt [B,H,hd,S] (all bf16).
// One wave computes a 16x64 strip.  Per k-chunk, all 10 b128 fragment loads
// are issued as one clause before the 4 WMMAs -> partial s_wait_loadcnt,
// single-buffered so nothing spills.
// ---------------------------------------------------------------------------
__global__ void k_qkv_gemm(const __bf16* __restrict__ X, const __bf16* __restrict__ Wt,
                           const float* __restrict__ bias,
                           __bf16* __restrict__ Q, __bf16* __restrict__ K,
                           __bf16* __restrict__ Vt) {
    const int wid     = blockIdx.x * 4 + (threadIdx.x >> 5);
    const int mi      = wid / (N3_ / 64);
    const int nj      = wid % (N3_ / 64);
    const int rowBase = mi * 16;
    const int colBase = nj * 64;
    const __bf16* Arow = X + rowBase * D_;

    v8f zero = {};
    v8f acc[4] = {zero, zero, zero, zero};

    for (int kb = 0; kb < D_; kb += 32) {
        __builtin_prefetch(Arow + kb + 128, 0, 3);
        // batch all fragment loads for this chunk (one load clause) ...
        v16bf a = load_a16(Arow, D_, kb);
        v16bf bf[4];
#pragma unroll
        for (int t = 0; t < 4; ++t) bf[t] = load_bt(Wt, D_, colBase + t * 16, kb);
        // ... then consume; later fragments stay in flight under early WMMAs
#pragma unroll
        for (int t = 0; t < 4; ++t) acc[t] = wmma_bf16(a, bf[t], acc[t]);
    }

    const int lane = threadIdx.x & 31;
    const int half = lane >> 4;
    const int nl   = lane & 15;
#pragma unroll
    for (int t = 0; t < 4; ++t) {
        const int col = colBase + t * 16 + nl;         // [0, 3072)
        const float bv = bias[col];
        const int h   = col / (3 * HD_);
        const int w   = col % (3 * HD_);
        const int sel = w >> 6;                        // 0:q 1:k 2:v
        const int d   = w & 63;
#pragma unroll
        for (int i = 0; i < 8; ++i) {
            const int row = rowBase + i + half * 8;    // [0, 4096)
            const int bb  = row >> 11;
            const int s   = row & (S_ - 1);
            const __bf16 val = f2bf(acc[t][i] + bv);
            if (sel == 0)      Q[(((bb * H_ + h) * S_) + s) * HD_ + d] = val;
            else if (sel == 1) K[(((bb * H_ + h) * S_) + s) * HD_ + d] = val;
            else               Vt[((bb * H_ + h) * HD_ + d) * S_ + s]  = val;
        }
    }
}

// ---------------------------------------------------------------------------
// Kernel 4: flash attention.  One wave owns one 16-query block of one (b,h).
// Online softmax; P re-striped C-layout -> A-layout through LDS.
// K fragments batch-loaded before the score WMMAs; V fragments issued right
// after the LDS P-stores so they overlap the ds-wait + ds-load, without
// being live across the whole softmax block (avoids spills).
// ---------------------------------------------------------------------------
__global__ void k_attention(const __bf16* __restrict__ Q, const __bf16* __restrict__ K,
                            const __bf16* __restrict__ Vt, __bf16* __restrict__ O) {
    __shared__ __align__(32) __bf16 lds_p[4][16][32];

    const int widx = threadIdx.x >> 5;
    const int wid  = blockIdx.x * 4 + widx;
    const int qb   = wid & (S_ / 16 - 1);        // 128 q-blocks
    const int h    = (wid >> 7) & (H_ - 1);
    const int b    = wid >> 11;

    const __bf16* Qp = Q  + ((b * H_ + h) * S_ + qb * 16) * HD_;
    const __bf16* Kp = K  + ((b * H_ + h) * S_) * HD_;
    const __bf16* Vp = Vt + ((b * H_ + h) * HD_) * S_;

    const v16bf qa0 = load_a16(Qp, HD_, 0);
    const v16bf qa1 = load_a16(Qp, HD_, 32);

    v8f zero = {};
    v8f acco[4] = {zero, zero, zero, zero};
    float mrun[8], lrun[8];
#pragma unroll
    for (int i = 0; i < 8; ++i) { mrun[i] = -INFINITY; lrun[i] = 0.0f; }

    const int lane = threadIdx.x & 31;
    const int half = lane >> 4;
    const int nl   = lane & 15;
    const float sc = 0.125f;                      // 1/sqrt(64)

    for (int kb = 0; kb < S_; kb += 32) {
        __builtin_prefetch(Kp + (kb + 32) * HD_, 0, 3);
        __builtin_prefetch(Vp + kb + 32, 0, 3);

        // batch K fragments (one clause), then score WMMAs
        v16bf kf[4];
        kf[0] = load_bt(Kp, HD_, kb, 0);
        kf[1] = load_bt(Kp, HD_, kb, 32);
        kf[2] = load_bt(Kp, HD_, kb + 16, 0);
        kf[3] = load_bt(Kp, HD_, kb + 16, 32);
        v8f s0 = zero, s1 = zero;
        s0 = wmma_bf16(qa0, kf[0], s0);
        s0 = wmma_bf16(qa1, kf[1], s0);
        s1 = wmma_bf16(qa0, kf[2], s1);
        s1 = wmma_bf16(qa1, kf[3], s1);

        float cf[8];
#pragma unroll
        for (int i = 0; i < 8; ++i) {
            const float a0 = s0[i] * sc;
            const float a1 = s1[i] * sc;
            const float tmax = rowmax16(fmaxf(a0, a1));
            const float mnew = fmaxf(mrun[i], tmax);
            cf[i] = __expf(mrun[i] - mnew);
            const float p0 = __expf(a0 - mnew);
            const float p1 = __expf(a1 - mnew);
            lrun[i] = lrun[i] * cf[i] + rowsum16(p0 + p1);
            mrun[i] = mnew;
            const int row = i + half * 8;
            lds_p[widx][row][nl]      = f2bf(p0);
            lds_p[widx][row][nl + 16] = f2bf(p1);
        }

        // issue V fragments now: they fly under the ds-wait + pa load
        v16bf vb[4];
#pragma unroll
        for (int t = 0; t < 4; ++t) vb[t] = load_bt(Vp, S_, t * 16, kb);

        // rescale running output while loads are in flight
#pragma unroll
        for (int t = 0; t < 4; ++t)
#pragma unroll
            for (int i = 0; i < 8; ++i) acco[t][i] *= cf[i];

        asm volatile("s_wait_dscnt 0" ::: "memory");   // P stores -> P loads (same wave)

        const v16bf pa = load_a16(&lds_p[widx][0][0], 32, 0);
#pragma unroll
        for (int t = 0; t < 4; ++t) acco[t] = wmma_bf16(pa, vb[t], acco[t]);
    }

    // normalize and store attn (bf16, [B*S][D] row-major) for output projection
#pragma unroll
    for (int t = 0; t < 4; ++t) {
#pragma unroll
        for (int i = 0; i < 8; ++i) {
            const int srow = qb * 16 + i + half * 8;
            const int col  = h * HD_ + t * 16 + nl;
            O[(b * S_ + srow) * D_ + col] = f2bf(acco[t][i] / lrun[i]);
        }
    }
}

// ---------------------------------------------------------------------------
// Kernel 5: output projection.  out(f32) = attn(bf16) @ W_out + b_out.
// Same batched-chunk scheme as k_qkv_gemm.
// ---------------------------------------------------------------------------
__global__ void k_out_gemm(const __bf16* __restrict__ A, const __bf16* __restrict__ Wt,
                           const float* __restrict__ bias, float* __restrict__ out) {
    const int wid     = blockIdx.x * 4 + (threadIdx.x >> 5);
    const int mi      = wid / (D_ / 64);
    const int nj      = wid % (D_ / 64);
    const int rowBase = mi * 16;
    const int colBase = nj * 64;
    const __bf16* Arow = A + rowBase * D_;

    v8f zero = {};
    v8f acc[4] = {zero, zero, zero, zero};

    for (int kb = 0; kb < D_; kb += 32) {
        __builtin_prefetch(Arow + kb + 128, 0, 3);
        v16bf a = load_a16(Arow, D_, kb);
        v16bf bf[4];
#pragma unroll
        for (int t = 0; t < 4; ++t) bf[t] = load_bt(Wt, D_, colBase + t * 16, kb);
#pragma unroll
        for (int t = 0; t < 4; ++t) acc[t] = wmma_bf16(a, bf[t], acc[t]);
    }

    const int lane = threadIdx.x & 31;
    const int half = lane >> 4;
    const int nl   = lane & 15;
#pragma unroll
    for (int t = 0; t < 4; ++t) {
        const int col = colBase + t * 16 + nl;
        const float bv = bias[col];
#pragma unroll
        for (int i = 0; i < 8; ++i) {
            const int row = rowBase + i + half * 8;
            out[row * D_ + col] = acc[t][i] + bv;
        }
    }
}

// ---------------------------------------------------------------------------
// Host launch
// ---------------------------------------------------------------------------
extern "C" void kernel_launch(void* const* d_in, const int* in_sizes, int n_in,
                              void* d_out, int out_size, void* d_ws, size_t ws_size,
                              hipStream_t stream) {
    const float* x     = (const float*)d_in[0];   // [B,S,D]
    const float* W_qkv = (const float*)d_in[1];   // [D, 3D]
    const float* b_qkv = (const float*)d_in[2];   // [3D]
    const float* W_out = (const float*)d_in[3];   // [D, D]
    const float* b_out = (const float*)d_in[4];   // [D]
    float* out = (float*)d_out;

    char* ws = (char*)d_ws;
    const size_t MB = 1024 * 1024;
    __bf16* Xbf    = (__bf16*)(ws + 0 * MB);      // 8 MB : [M, D]
    __bf16* Wqkv_t = (__bf16*)(ws + 8 * MB);      // 6 MB : [3D, D]
    __bf16* Wout_t = (__bf16*)(ws + 14 * MB);     // 2 MB : [D, D]
    __bf16* Qb     = (__bf16*)(ws + 16 * MB);     // 8 MB : [B,H,S,hd]
    __bf16* Kb     = (__bf16*)(ws + 24 * MB);     // 8 MB : [B,H,S,hd]
    __bf16* Vtb    = (__bf16*)(ws + 32 * MB);     // 8 MB : [B,H,hd,S]
    __bf16* Attn   = (__bf16*)(ws + 40 * MB);     // 8 MB : [M, D]

    // 1) precision casts / weight transposes
    k_cast_bf16<<<(M_ * D_) / 256, 256, 0, stream>>>(x, Xbf, M_ * D_);
    k_transpose_bf16<<<(D_ * N3_) / 256, 256, 0, stream>>>(W_qkv, Wqkv_t, D_, N3_);
    k_transpose_bf16<<<(D_ * D_) / 256, 256, 0, stream>>>(W_out, Wout_t, D_, D_);

    // 2) QKV projection: (M/16)*(3072/64) = 256*48 = 12288 waves, 4 waves/block
    k_qkv_gemm<<<12288 / 4, 128, 0, stream>>>(Xbf, Wqkv_t, b_qkv, Qb, Kb, Vtb);

    // 3) attention: B*H*(S/16) = 4096 waves, 4 waves/block
    k_attention<<<4096 / 4, 128, 0, stream>>>(Qb, Kb, Vtb, Attn);

    // 4) output projection: (M/16)*(D/64) = 256*16 = 4096 waves
    k_out_gemm<<<4096 / 4, 128, 0, stream>>>(Attn, Wout_t, b_out, out);
}
